// QueryDecoder_63771674411398
// MI455X (gfx1250) — matrix operand order repaired and checked by reading, not compiled, not run on hardware
//
#include <hip/hip_runtime.h>

// ---------------- model dimensions ----------------
#define kB    8
#define kN    4096
#define kNQ   100
#define kNQP  128
#define kD    256
#define kH    8
#define kHD   32
#define kIN   32
#define kHID  1024
#define kL    6
#define kNCLS 19
#define LOGBETA (-0.1053605156578263f)   // log(0.9)

typedef __bf16 bf16;
typedef __attribute__((ext_vector_type(16))) bf16  bf16x16;
typedef __attribute__((ext_vector_type(8)))  bf16  bf16x8;
typedef __attribute__((ext_vector_type(8)))  float floatx8;

// ---------------- device helpers ----------------
__device__ inline floatx8 fzero8() {
  floatx8 z;
#pragma unroll
  for (int i = 0; i < 8; ++i) z[i] = 0.f;
  return z;
}
__device__ inline bf16x16 bpack(bf16x8 lo, bf16x8 hi) {
  union { bf16x16 v; bf16x8 h[2]; } u;
  u.h[0] = lo; u.h[1] = hi;
  return u.v;
}
__device__ inline floatx8 wmma_bf(bf16x16 a, bf16x16 b, floatx8 c) {
  // D(f32 16x16) = A(16x32 bf16) * B(32x16 bf16) + C
  return __builtin_amdgcn_wmma_f32_16x16x32_bf16(false, a, false, b, (short)0, c, false, false);
}

// ---------------- generic batched WMMA GEMM ----------------
// C[M,N] = act(A[M,K] @ Bw[N,K]^T + bias [+ resid])
// Block tile 128x128, 8 waves as 4(M) x 2(N); wave tile 32x64 -> 8 WMMA per K-step.
// All fragment loads of one K-step issue before the WMMA burst (independent registers),
// so one clause of loads covers 8 WMMAs. B rows clamped; cols >= N never stored.
__global__ __launch_bounds__(256) void gemm_bf16_k(
    const bf16* __restrict__ A, long long sA,
    const bf16* __restrict__ Bw, long long sB,
    const float* __restrict__ bias,
    const float* __restrict__ resid,
    float* __restrict__ Cf, bf16* __restrict__ Cb, long long sC,
    int M, int N, int K, int act) {
  int bz = blockIdx.z;
  const bf16* Ab = A + (long long)bz * sA;
  const bf16* Bb = Bw + (long long)bz * sB;
  int tn = blockIdx.x * 128, tm = blockIdx.y * 128;
  int w  = threadIdx.x >> 5;
  long long row0 = tm + (w & 3) * 32;
  int n0 = tn + (w >> 2) * 64;
  int lane = threadIdx.x & 31;
  int cl = lane & 15;

  // loop-invariant fragment base pointers
  const bf16* ap0 = Ab + (row0 + (lane & 15)) * (long long)K + ((lane & 16) ? 8 : 0);
  const bf16* ap1 = ap0 + (long long)16 * K;
  const bf16* bp[4];
#pragma unroll
  for (int j = 0; j < 4; ++j) {
    int n = n0 + j * 16 + cl;
    n = (n < N) ? n : (N - 1);           // clamp instead of diverge
    bp[j] = Bb + (long long)n * K + ((lane & 16) ? 8 : 0);
  }

  floatx8 c[2][4];
#pragma unroll
  for (int i = 0; i < 2; ++i)
#pragma unroll
    for (int j = 0; j < 4; ++j) c[i][j] = fzero8();

  for (int k = 0; k < K; k += 32) {
    // issue ALL fragment loads for this K-step first (independent destinations)
    bf16x16 a0 = bpack(*(const bf16x8*)(ap0 + k), *(const bf16x8*)(ap0 + k + 16));
    bf16x16 a1 = bpack(*(const bf16x8*)(ap1 + k), *(const bf16x8*)(ap1 + k + 16));
    bf16x16 bfr[4];
#pragma unroll
    for (int j = 0; j < 4; ++j) bfr[j] = bpack(*(const bf16x8*)(bp[j] + k), *(const bf16x8*)(bp[j] + k + 16));
    if (k + 32 < K) {
      __builtin_prefetch(ap0 + k + 32, 0, 1);
      __builtin_prefetch(ap1 + k + 32, 0, 1);
    }
    // 8-WMMA burst behind a single wait
#pragma unroll
    for (int j = 0; j < 4; ++j) {
      c[0][j] = wmma_bf(a0, bfr[j], c[0][j]);
      c[1][j] = wmma_bf(a1, bfr[j], c[1][j]);
    }
  }

  int rb = (lane & 16) ? 8 : 0;
#pragma unroll
  for (int i = 0; i < 2; ++i)
#pragma unroll
    for (int j = 0; j < 4; ++j) {
      int col = n0 + j * 16 + cl;
      if (col >= N) continue;
      float bv = bias ? bias[col] : 0.f;
#pragma unroll
      for (int r = 0; r < 8; ++r) {
        long long row = row0 + i * 16 + rb + r;
        float v = c[i][j][r] + bv;
        long long oi = (long long)bz * sC + row * (long long)N + col;
        if (resid) v += resid[oi];
        if (act == 1) v = fmaxf(v, 0.f);
        if (Cf) Cf[oi] = v;
        if (Cb) Cb[oi] = (bf16)v;
      }
    }
}

// ---------------- fused flash attention ----------------
// grid (H, B), 8 waves, 128 padded queries, head dim 32, NK multiple of 128
__global__ __launch_bounds__(256) void flash_attn_k(
    const bf16* __restrict__ Q,      // (B*kNQP, kD)
    const bf16* __restrict__ Kp,     // (B*NK, kD)
    const bf16* __restrict__ Vp,     // (B*NK, kD)
    const float* __restrict__ biasP, // (B, kNQP, NK) or null
    bf16* __restrict__ O,            // (B*kNQP, kD)
    int NK) {
  __shared__ bf16 Ks[128][32];       // K tile, row-major
  __shared__ bf16 Vt[32][128];       // V tile, transposed (d, key)
  __shared__ bf16 Ps[8][16][128];    // per-wave P tile for layout transpose

  int h = blockIdx.x, b = blockIdx.y;
  int tid = threadIdx.x, w = tid >> 5, lane = tid & 31;
  int m = lane & 15;
  int hlf = (lane & 16) ? 1 : 0;
  int rb = hlf * 8, cl = m;
  int qrow0 = b * kNQP + w * 16;

  bf16x16 aq;
  {
    int kb = hlf ? 8 : 0;
    const bf16* p = Q + (long long)(qrow0 + m) * kD + h * kHD + kb;
    aq = bpack(*(const bf16x8*)p, *(const bf16x8*)(p + 16));
  }
  float mrow[8], lrow[8];
#pragma unroll
  for (int r = 0; r < 8; ++r) { mrow[r] = -3.0e38f; lrow[r] = 0.f; }
  floatx8 o0 = fzero8(), o1 = fzero8();
  const float scale = 0.17677669529663687f;  // 1/sqrt(32)

  int ntiles = NK >> 7;
  for (int kt = 0; kt < ntiles; ++kt) {
    __syncthreads();
    {  // cooperative staging of K and V tiles (stream-once -> non-temporal)
      int i = tid >> 1, hh = tid & 1;
      long long gr = ((long long)b * NK + (long long)kt * 128 + i) * kD + h * kHD + hh * 16;
      bf16x16 vK = __builtin_nontemporal_load((const bf16x16*)(Kp + gr));
      *(bf16x16*)&Ks[i][hh * 16] = vK;
      bf16x16 vV = __builtin_nontemporal_load((const bf16x16*)(Vp + gr));
#pragma unroll
      for (int d2 = 0; d2 < 16; ++d2) Vt[hh * 16 + d2][i] = vV[d2];
    }
    __syncthreads();

    floatx8 s[8];
#pragma unroll
    for (int j = 0; j < 8; ++j) {
      bf16x16 bk = bpack(*(const bf16x8*)&Ks[j * 16 + cl][hlf ? 8 : 0], *(const bf16x8*)&Ks[j * 16 + cl][hlf ? 24 : 16]);
      s[j] = wmma_bf(aq, bk, fzero8());
    }
    float pm[8];
#pragma unroll
    for (int r = 0; r < 8; ++r) pm[r] = -3.0e38f;
#pragma unroll
    for (int j = 0; j < 8; ++j)
#pragma unroll
      for (int r = 0; r < 8; ++r) {
        float v = s[j][r] * scale;
        if (biasP) {
          int row = w * 16 + rb + r;
          int col = kt * 128 + j * 16 + cl;
          v += biasP[((long long)b * kNQP + row) * (long long)NK + col];
        }
        s[j][r] = v;
        pm[r] = fmaxf(pm[r], v);
      }
#pragma unroll
    for (int off = 8; off >= 1; off >>= 1)
#pragma unroll
      for (int r = 0; r < 8; ++r) pm[r] = fmaxf(pm[r], __shfl_xor(pm[r], off, 16));
    float corr[8];
#pragma unroll
    for (int r = 0; r < 8; ++r) {
      float mn = fmaxf(mrow[r], pm[r]);
      corr[r] = __expf(mrow[r] - mn);
      mrow[r] = mn;
    }
    float psum[8];
#pragma unroll
    for (int r = 0; r < 8; ++r) psum[r] = 0.f;
#pragma unroll
    for (int j = 0; j < 8; ++j)
#pragma unroll
      for (int r = 0; r < 8; ++r) {
        float p = __expf(s[j][r] - mrow[r]);
        psum[r] += p;
        Ps[w][rb + r][j * 16 + cl] = (bf16)p;
      }
#pragma unroll
    for (int off = 8; off >= 1; off >>= 1)
#pragma unroll
      for (int r = 0; r < 8; ++r) psum[r] += __shfl_xor(psum[r], off, 16);
#pragma unroll
    for (int r = 0; r < 8; ++r) {
      lrow[r] = lrow[r] * corr[r] + psum[r];
      o0[r] *= corr[r];
      o1[r] *= corr[r];
    }
    __syncthreads();
#pragma unroll
    for (int ks = 0; ks < 4; ++ks) {
      int ab = hlf ? 16 : 0;
      const bf16* pp = &Ps[w][m][ks * 32 + ab];
      bf16x16 aP = *(const bf16x16*)pp;
      int kb16 = hlf ? 16 : 0;
      bf16x16 bv0 = *(const bf16x16*)&Vt[cl][ks * 32 + kb16];
      bf16x16 bv1 = *(const bf16x16*)&Vt[16 + cl][ks * 32 + kb16];
      o0 = wmma_bf(aP, bv0, o0);
      o1 = wmma_bf(aP, bv1, o1);
    }
  }
#pragma unroll
  for (int r = 0; r < 8; ++r) {
    float inv = 1.f / lrow[r];
    long long row = qrow0 + rb + r;
    O[row * (long long)kD + h * kHD + cl]      = (bf16)(o0[r] * inv);
    O[row * (long long)kD + h * kHD + 16 + cl] = (bf16)(o1[r] * inv);
  }
}

// ---------------- layernorm over D=256, one block per row ----------------
__global__ __launch_bounds__(256) void ln256_k(const float* __restrict__ X,
                                               const float* __restrict__ g,
                                               const float* __restrict__ be,
                                               float* __restrict__ Yf, bf16* __restrict__ Yb,
                                               int relu_after) {
  long long row = blockIdx.x;
  int t = threadIdx.x;
  float v = X[row * kD + t];
  __shared__ float s1[256], s2[256];
  s1[t] = v; s2[t] = v * v;
  __syncthreads();
  for (int s = 128; s > 0; s >>= 1) {
    if (t < s) { s1[t] += s1[t + s]; s2[t] += s2[t + s]; }
    __syncthreads();
  }
  float mean = s1[0] * (1.f / kD);
  float var  = s2[0] * (1.f / kD) - mean * mean;
  float y = (v - mean) * rsqrtf(var + 1e-5f) * g[t] + be[t];
  if (relu_after) y = fmaxf(y, 0.f);
  if (Yf) Yf[row * kD + t] = y;
  if (Yb) Yb[row * kD + t] = (bf16)y;
}

// ---------------- small utility kernels ----------------
__global__ void f2b_k(const float* __restrict__ s, bf16* __restrict__ d, long long n) {
  long long i = (long long)blockIdx.x * 256 + threadIdx.x;
  if (i < n) d[i] = (bf16)s[i];
}
__global__ void xsmean_k(const float* __restrict__ xm, float* __restrict__ xs) {
  int i = blockIdx.x * 256 + threadIdx.x;
  if (i >= kB * kN) return;
  float s = 0.f;
  const float* r = xm + (long long)i * kIN;
  for (int j = 0; j < kIN; ++j) s += r[j];
  xs[i] = s * (1.f / kIN);
}
// per-batch select NQ smallest (ties -> lowest index), matches top_k(-xs)
__global__ __launch_bounds__(256) void topk_k(const float* __restrict__ xs, int* __restrict__ idx) {
  int b = blockIdx.x, t = threadIdx.x;
  __shared__ float vals[kN];
  __shared__ float rv[256];
  __shared__ int ri[256];
  for (int i = t; i < kN; i += 256) vals[i] = xs[b * kN + i];
  __syncthreads();
  for (int s = 0; s < kNQ; ++s) {
    float bv = 3e38f; int bi = 0;
    for (int i = t; i < kN; i += 256) {
      float v = vals[i];
      if (v < bv) { bv = v; bi = i; }
    }
    rv[t] = bv; ri[t] = bi;
    __syncthreads();
    for (int s2 = 128; s2 > 0; s2 >>= 1) {
      if (t < s2) {
        if (rv[t + s2] < rv[t] || (rv[t + s2] == rv[t] && ri[t + s2] < ri[t])) {
          rv[t] = rv[t + s2]; ri[t] = ri[t + s2];
        }
      }
      __syncthreads();
    }
    if (t == 0) { idx[b * kNQ + s] = ri[0]; vals[ri[0]] = 3e38f; }
    __syncthreads();
  }
}
__global__ __launch_bounds__(256) void gather_k(const float* __restrict__ inst,
                                                const float* __restrict__ coords,
                                                const int* __restrict__ idx,
                                                float* __restrict__ qs_f, bf16* __restrict__ qs_b,
                                                float* __restrict__ qc) {
  int blk = blockIdx.x, b = blk >> 7, q = blk & 127, t = threadIdx.x;
  bool valid = q < kNQ;
  int src = valid ? idx[b * kNQ + q] : 0;
  float v = valid ? inst[((long long)b * kN + src) * kD + t] : 0.f;
  qs_f[(long long)blk * kD + t] = v;
  qs_b[(long long)blk * kD + t] = (bf16)v;
  if (t < 3) qc[blk * 3 + t] = valid ? coords[((long long)b * kN + src) * 3 + t] : 1.0e9f;
}
__global__ void rowscale_k(const float* __restrict__ X, const float* __restrict__ srow,
                           bf16* __restrict__ Y, long long n) {
  long long i = (long long)blockIdx.x * 256 + threadIdx.x;
  if (i >= n) return;
  Y[i] = (bf16)(X[i] * srow[i >> 8]);   // row stride kD=256
}
__global__ void cwbuild_k(const float* __restrict__ qc, const float* __restrict__ coords,
                          float* __restrict__ cw) {
  long long i = (long long)blockIdx.x * 256 + threadIdx.x;
  if (i >= (long long)kB * kNQP * kN) return;
  int k = (int)(i & (kN - 1));
  long long t2 = i >> 12;
  int q = (int)(t2 & (kNQP - 1));
  int b = (int)(t2 >> 7);
  const float* a  = &qc[(b * kNQP + q) * 3];
  const float* c2 = &coords[((long long)b * kN + k) * 3];
  float d = fmaxf(fabsf(a[0] - c2[0]), fmaxf(fabsf(a[1] - c2[1]), fabsf(a[2] - c2[2])));
  cw[i] = __expf(d * LOGBETA);
}
// rank-1 SVD factor by power iteration: u (B,128), sv = s0*v (B,N)
__global__ __launch_bounds__(256) void powit_k(const float* __restrict__ cw,
                                               float* __restrict__ u, float* __restrict__ sv) {
  int b = blockIdx.x, t = threadIdx.x;
  __shared__ float vs[kN];
  __shared__ float us[kNQP];
  __shared__ float red[256];
  __shared__ float snorm;
  const float* C = cw + (long long)b * kNQP * kN;
  for (int i = t; i < kN; i += 256) vs[i] = 1.f;
  __syncthreads();
  for (int it = 0; it < 10; ++it) {
    if (t < kNQP) {
      float s = 0.f;
      const float* row = C + (long long)t * kN;
      for (int k2 = 0; k2 < kN; ++k2) s += row[k2] * vs[k2];
      us[t] = s;
    }
    __syncthreads();
    red[t] = (t < kNQP) ? us[t] * us[t] : 0.f;
    __syncthreads();
    for (int s2 = 128; s2 > 0; s2 >>= 1) { if (t < s2) red[t] += red[t + s2]; __syncthreads(); }
    if (t == 0) snorm = rsqrtf(red[0] + 1e-30f);
    __syncthreads();
    if (t < kNQP) us[t] *= snorm;
    __syncthreads();
    for (int c2 = t; c2 < kN; c2 += 256) {
      float s = 0.f;
      for (int r2 = 0; r2 < kNQP; ++r2) s += C[(long long)r2 * kN + c2] * us[r2];
      vs[c2] = s;
    }
    __syncthreads();
    float q = 0.f;
    for (int c2 = t; c2 < kN; c2 += 256) q += vs[c2] * vs[c2];
    red[t] = q;
    __syncthreads();
    for (int s2 = 128; s2 > 0; s2 >>= 1) { if (t < s2) red[t] += red[t + s2]; __syncthreads(); }
    if (t == 0) snorm = rsqrtf(red[0] + 1e-30f);
    __syncthreads();
    for (int c2 = t; c2 < kN; c2 += 256) vs[c2] *= snorm;
    __syncthreads();
  }
  if (t < kNQP) {
    float s = 0.f;
    const float* row = C + (long long)t * kN;
    for (int k2 = 0; k2 < kN; ++k2) s += row[k2] * vs[k2];
    us[t] = s;
  }
  __syncthreads();
  red[t] = (t < kNQP) ? us[t] * us[t] : 0.f;
  __syncthreads();
  for (int s2 = 128; s2 > 0; s2 >>= 1) { if (t < s2) red[t] += red[t + s2]; __syncthreads(); }
  if (t == 0) snorm = sqrtf(red[0]) + 1e-30f;   // s0
  __syncthreads();
  if (t < kNQP) u[b * kNQP + t] = us[t] / snorm;
  for (int c2 = t; c2 < kN; c2 += 256) sv[(long long)b * kN + c2] = vs[c2] * snorm;
}
// self-attn bool mask -> additive bias (B,128,128)
__global__ __launch_bounds__(256) void selfbias_k(const float* __restrict__ qc,
                                                  float* __restrict__ sb) {
  int b = blockIdx.x, t = threadIdx.x;
  __shared__ float red[256];
  __shared__ float meanS;
  float part = 0.f;
  for (int i = t; i < kNQ * kNQ; i += 256) {
    int q = i / kNQ, k2 = i % kNQ;
    const float* A  = &qc[(b * kNQP + q) * 3];
    const float* C2 = &qc[(b * kNQP + k2) * 3];
    float d = fmaxf(fabsf(A[0] - C2[0]), fmaxf(fabsf(A[1] - C2[1]), fabsf(A[2] - C2[2])));
    part += __expf(d * LOGBETA);
  }
  red[t] = part;
  __syncthreads();
  for (int s2 = 128; s2 > 0; s2 >>= 1) { if (t < s2) red[t] += red[t + s2]; __syncthreads(); }
  if (t == 0) meanS = red[0] / (float)(kNQ * kNQ);
  __syncthreads();
  for (int i = t; i < kNQP * kNQP; i += 256) {
    int q = i >> 7, k2 = i & 127;
    float v;
    if (k2 >= kNQ) v = -1e9f;
    else {
      const float* A  = &qc[(b * kNQP + q) * 3];
      const float* C2 = &qc[(b * kNQP + k2) * 3];
      float d = fmaxf(fabsf(A[0] - C2[0]), fmaxf(fabsf(A[1] - C2[1]), fabsf(A[2] - C2[2])));
      v = (__expf(d * LOGBETA) < meanS) ? -1e9f : 0.f;
    }
    sb[((long long)b * kNQP + q) * kNQP + k2] = v;
  }
}
__global__ void pack_k(const float* __restrict__ labels_pad, const float* __restrict__ scores_pad,
                       const float* __restrict__ masks_pad, float* __restrict__ out) {
  const long long NL = (long long)kB * kNQ * kNCLS;
  const long long NS = (long long)kB * kNQ;
  const long long NM = (long long)kB * kNQ * kN;
  long long i = (long long)blockIdx.x * 256 + threadIdx.x;
  if (i >= NL + NS + NM) return;
  if (i < NL) {
    long long t2 = i; int c = (int)(t2 % kNCLS); t2 /= kNCLS;
    int q = (int)(t2 % kNQ), b = (int)(t2 / kNQ);
    out[i] = labels_pad[((long long)b * kNQP + q) * kNCLS + c];
  } else if (i < NL + NS) {
    long long t2 = i - NL; int q = (int)(t2 % kNQ), b = (int)(t2 / kNQ);
    out[i] = scores_pad[b * kNQP + q];
  } else {
    long long t2 = i - NL - NS; int n = (int)(t2 % kN); t2 /= kN;
    int q = (int)(t2 % kNQ), b = (int)(t2 / kNQ);
    out[i] = masks_pad[((long long)b * kNQP + q) * kN + n];
  }
}

// ---------------- host orchestration ----------------
extern "C" void kernel_launch(void* const* d_in, const int* in_sizes, int n_in,
                              void* d_out, int out_size, void* d_ws, size_t ws_size,
                              hipStream_t stream) {
  (void)in_sizes; (void)n_in; (void)out_size; (void)ws_size;
  const float* x        = (const float*)d_in[0];
  const float* x_mar    = (const float*)d_in[1];
  const float* sp       = (const float*)d_in[2];
  const float* inproj_w = (const float*)d_in[4];
  const float* inproj_b = (const float*)d_in[5];
  const float* inln_g   = (const float*)d_in[6];
  const float* inln_b   = (const float*)d_in[7];
  const float* xmask_w1 = (const float*)d_in[8];
  const float* xmask_b1 = (const float*)d_in[9];
  const float* xmask_w2 = (const float*)d_in[10];
  const float* xmask_b2 = (const float*)d_in[11];
  const float* gq_w1    = (const float*)d_in[12];
  const float* gq_b1    = (const float*)d_in[13];
  const float* gq_w2    = (const float*)d_in[14];
  const float* gq_b2    = (const float*)d_in[15];
  const float* gq_g     = (const float*)d_in[16];
  const float* gq_bb    = (const float*)d_in[17];
  const float* cross_w  = (const float*)d_in[18];
  const float* cross_b  = (const float*)d_in[19];
  const float* cross_ow = (const float*)d_in[20];
  const float* cross_ob = (const float*)d_in[21];
  const float* self_w   = (const float*)d_in[22];
  const float* self_b   = (const float*)d_in[23];
  const float* self_ow  = (const float*)d_in[24];
  const float* self_ob  = (const float*)d_in[25];
  const float* self_g   = (const float*)d_in[26];
  const float* self_bb  = (const float*)d_in[27];
  const float* ffn_w1   = (const float*)d_in[28];
  const float* ffn_b1   = (const float*)d_in[29];
  const float* ffn_w2   = (const float*)d_in[30];
  const float* ffn_b2   = (const float*)d_in[31];
  const float* ffn_g    = (const float*)d_in[32];
  const float* ffn_bb   = (const float*)d_in[33];
  const float* outln_g  = (const float*)d_in[34];
  const float* outln_b  = (const float*)d_in[35];
  const float* cls_w1   = (const float*)d_in[36];
  const float* cls_b1   = (const float*)d_in[37];
  const float* cls_w2   = (const float*)d_in[38];
  const float* cls_b2   = (const float*)d_in[39];
  const float* score_w1 = (const float*)d_in[40];
  const float* score_b1 = (const float*)d_in[41];
  const float* score_w2 = (const float*)d_in[42];
  const float* score_b2 = (const float*)d_in[43];

  char* base = (char*)d_ws;
  size_t off = 0;
  auto alloc = [&](size_t bytes) -> void* {
    void* p = base + off;
    off += (bytes + 255) & ~(size_t)255;
    return p;
  };
  const long long TOK = (long long)kB * kN;       // 32768
  const long long QTOK = (long long)kB * kNQP;    // 1024

  bf16* wb_inproj = (bf16*)alloc(kD * kIN * 2);
  bf16* wb_xm1    = (bf16*)alloc(kD * kIN * 2);
  bf16* wb_xm2    = (bf16*)alloc(kD * kD * 2);
  bf16* wb_gq1    = (bf16*)alloc(kHID * kD * 2);
  bf16* wb_gq2    = (bf16*)alloc(kD * kHID * 2);
  bf16* wb_crw    = (bf16*)alloc((size_t)kL * 3 * kD * kD * 2);
  bf16* wb_crow   = (bf16*)alloc((size_t)kL * kD * kD * 2);
  bf16* wb_sw     = (bf16*)alloc((size_t)kL * 3 * kD * kD * 2);
  bf16* wb_sow    = (bf16*)alloc((size_t)kL * kD * kD * 2);
  bf16* wb_f1     = (bf16*)alloc((size_t)kL * kHID * kD * 2);
  bf16* wb_f2     = (bf16*)alloc((size_t)kL * kD * kHID * 2);
  bf16* wb_c1     = (bf16*)alloc(kD * kD * 2);
  bf16* wb_c2     = (bf16*)alloc(kNCLS * kD * 2);
  bf16* wb_s1     = (bf16*)alloc(kD * kD * 2);
  bf16* wb_s2     = (bf16*)alloc(kD * 2);
  bf16* x_bf      = (bf16*)alloc(TOK * kIN * 2);
  // aliased region: t0 (pre-layer) overlaps cw (layers) + masks_pad (heads)
  char* regionA   = (char*)alloc((size_t)34 * 1024 * 1024);
  float* t0_f32     = (float*)regionA;                                  // 32 MB
  float* cw_f32     = (float*)regionA;                                  // 16.8 MB
  float* masks_pad  = (float*)(regionA + (size_t)17 * 1024 * 1024);     // 16.8 MB
  float* inst_f32 = (float*)alloc(TOK * kD * 4);
  bf16*  inst_bf  = (bf16*)alloc(TOK * kD * 2);
  bf16*  mfe_bf   = (bf16*)alloc(TOK * kD * 2);
  bf16*  mk_bf    = (bf16*)alloc(TOK * kD * 2);   // xmask hidden, then kw per layer
  bf16*  kp_bf    = (bf16*)alloc(TOK * kD * 2);
  bf16*  vp_bf    = (bf16*)alloc(TOK * kD * 2);
  float* xs_f     = (float*)alloc(TOK * 4);
  int*   idx_i    = (int*)alloc(kB * kNQ * 4);
  float* qs_f     = (float*)alloc(QTOK * kD * 4);
  bf16*  qs_bf    = (bf16*)alloc(QTOK * kD * 2);
  float* qc_f     = (float*)alloc(QTOK * 3 * 4);
  float* u_f      = (float*)alloc(QTOK * 4);
  float* sv_f     = (float*)alloc(TOK * 4);
  float* q_f      = (float*)alloc(QTOK * kD * 4);
  bf16*  q_bf     = (bf16*)alloc(QTOK * kD * 2);
  float* t1_f     = (float*)alloc(QTOK * kD * 4);
  bf16*  hid_bf   = (bf16*)alloc(QTOK * kHID * 2);
  bf16*  qw_bf    = (bf16*)alloc(QTOK * kD * 2);  // also reused as qn_bf at the end
  bf16*  qp_bf    = (bf16*)alloc(QTOK * kD * 2);
  bf16*  kq_bf    = (bf16*)alloc(QTOK * kD * 2);
  bf16*  vq_bf    = (bf16*)alloc(QTOK * kD * 2);
  bf16*  o_bf     = (bf16*)alloc(QTOK * kD * 2);
  float* sbias    = (float*)alloc((size_t)kB * kNQP * kNQP * 4);
  bf16*  h1_bf    = (bf16*)alloc(QTOK * kD * 2);
  float* labels_pad = (float*)alloc(QTOK * kNCLS * 4);
  float* scores_pad = (float*)alloc(QTOK * 4);

  auto cvt = [&](const float* s, bf16* d, long long n) {
    f2b_k<<<dim3((unsigned)((n + 255) / 256)), dim3(256), 0, stream>>>(s, d, n);
  };
  auto gemm = [&](const bf16* A, long long sA, const bf16* Bw, long long sB,
                  const float* bias, const float* resid, float* Cf, bf16* Cb,
                  long long sC, int M, int N, int K, int act, int batch) {
    dim3 g((N + 127) / 128, M / 128, batch);
    gemm_bf16_k<<<g, dim3(256), 0, stream>>>(A, sA, Bw, sB, bias, resid, Cf, Cb, sC, M, N, K, act);
  };
  auto ln = [&](int rows, const float* X, const float* g, const float* be,
                float* Yf, bf16* Yb, int relu) {
    ln256_k<<<dim3((unsigned)rows), dim3(256), 0, stream>>>(X, g, be, Yf, Yb, relu);
  };

  // ---- weight / input conversion to bf16 ----
  cvt(x, x_bf, TOK * kIN);
  cvt(inproj_w, wb_inproj, kD * kIN);
  cvt(xmask_w1, wb_xm1, kD * kIN);
  cvt(xmask_w2, wb_xm2, kD * kD);
  cvt(gq_w1, wb_gq1, (long long)kHID * kD);
  cvt(gq_w2, wb_gq2, (long long)kD * kHID);
  cvt(cross_w, wb_crw, (long long)kL * 3 * kD * kD);
  cvt(cross_ow, wb_crow, (long long)kL * kD * kD);
  cvt(self_w, wb_sw, (long long)kL * 3 * kD * kD);
  cvt(self_ow, wb_sow, (long long)kL * kD * kD);
  cvt(ffn_w1, wb_f1, (long long)kL * kHID * kD);
  cvt(ffn_w2, wb_f2, (long long)kL * kD * kHID);
  cvt(cls_w1, wb_c1, kD * kD);
  cvt(cls_w2, wb_c2, kNCLS * kD);
  cvt(score_w1, wb_s1, kD * kD);
  cvt(score_w2, wb_s2, kD);

  // ---- inst = relu(LN(x @ inproj^T + b)) ----
  gemm(x_bf, 0, wb_inproj, 0, inproj_b, nullptr, t0_f32, nullptr, 0, (int)TOK, kD, kIN, 0, 1);
  ln((int)TOK, t0_f32, inln_g, inln_b, inst_f32, inst_bf, 1);
  // ---- mfe ----
  gemm(x_bf, 0, wb_xm1, 0, xmask_b1, nullptr, nullptr, mk_bf, 0, (int)TOK, kD, kIN, 1, 1);
  gemm(mk_bf, 0, wb_xm2, 0, xmask_b2, nullptr, nullptr, mfe_bf, 0, (int)TOK, kD, kD, 0, 1);
  // ---- query sampling + init FFN ----
  xsmean_k<<<dim3((unsigned)((TOK + 255) / 256)), dim3(256), 0, stream>>>(x_mar, xs_f);
  topk_k<<<dim3(kB), dim3(256), 0, stream>>>(xs_f, idx_i);
  gather_k<<<dim3((unsigned)QTOK), dim3(256), 0, stream>>>(inst_f32, sp, idx_i, qs_f, qs_bf, qc_f);
  gemm(qs_bf, 0, wb_gq1, 0, gq_b1, nullptr, nullptr, hid_bf, 0, (int)QTOK, kHID, kD, 1, 1);
  gemm(hid_bf, 0, wb_gq2, 0, gq_b2, qs_f, t1_f, nullptr, 0, (int)QTOK, kD, kHID, 0, 1);
  ln((int)QTOK, t1_f, gq_g, gq_bb, q_f, q_bf, 0);

  // ---- decoder layers ----
  for (int l = 0; l < kL; ++l) {
    cwbuild_k<<<dim3((unsigned)(((long long)kB * kNQP * kN) / 256)), dim3(256), 0, stream>>>(qc_f, sp, cw_f32);
    powit_k<<<dim3(kB), dim3(256), 0, stream>>>(cw_f32, u_f, sv_f);
    rowscale_k<<<dim3((unsigned)(QTOK * kD / 256)), dim3(256), 0, stream>>>(q_f, u_f, qw_bf, QTOK * kD);
    rowscale_k<<<dim3((unsigned)(TOK * kD / 256)), dim3(256), 0, stream>>>(inst_f32, sv_f, mk_bf, TOK * kD);

    const bf16* Wl = wb_crw + (long long)l * 3 * kD * kD;
    const float* bl = cross_b + (long long)l * 3 * kD;
    gemm(qw_bf, 0, Wl, 0, bl, nullptr, nullptr, qp_bf, 0, (int)QTOK, kD, kD, 0, 1);
    gemm(mk_bf, 0, Wl + (long long)kD * kD, 0, bl + kD, nullptr, nullptr, kp_bf, 0, (int)TOK, kD, kD, 0, 1);
    gemm(inst_bf, 0, Wl + (long long)2 * kD * kD, 0, bl + 2 * kD, nullptr, nullptr, vp_bf, 0, (int)TOK, kD, kD, 0, 1);
    flash_attn_k<<<dim3(kH, kB), dim3(256), 0, stream>>>(qp_bf, kp_bf, vp_bf, nullptr, o_bf, kN);
    gemm(o_bf, 0, wb_crow + (long long)l * kD * kD, 0, cross_ob + (long long)l * kD,
         q_f, q_f, q_bf, 0, (int)QTOK, kD, kD, 0, 1);

    selfbias_k<<<dim3(kB), dim3(256), 0, stream>>>(qc_f, sbias);
    const bf16* Sl = wb_sw + (long long)l * 3 * kD * kD;
    const float* sb2 = self_b + (long long)l * 3 * kD;
    gemm(q_bf, 0, Sl, 0, sb2, nullptr, nullptr, qp_bf, 0, (int)QTOK, kD, kD, 0, 1);
    gemm(q_bf, 0, Sl + (long long)kD * kD, 0, sb2 + kD, nullptr, nullptr, kq_bf, 0, (int)QTOK, kD, kD, 0, 1);
    gemm(q_bf, 0, Sl + (long long)2 * kD * kD, 0, sb2 + 2 * kD, nullptr, nullptr, vq_bf, 0, (int)QTOK, kD, kD, 0, 1);
    flash_attn_k<<<dim3(kH, kB), dim3(256), 0, stream>>>(qp_bf, kq_bf, vq_bf, sbias, o_bf, kNQP);
    gemm(o_bf, 0, wb_sow + (long long)l * kD * kD, 0, self_ob + (long long)l * kD,
         q_f, t1_f, nullptr, 0, (int)QTOK, kD, kD, 0, 1);
    ln((int)QTOK, t1_f, self_g + (long long)l * kD, self_bb + (long long)l * kD, q_f, q_bf, 0);

    gemm(q_bf, 0, wb_f1 + (long long)l * kHID * kD, 0, ffn_b1 + (long long)l * kHID,
         nullptr, nullptr, hid_bf, 0, (int)QTOK, kHID, kD, 1, 1);
    gemm(hid_bf, 0, wb_f2 + (long long)l * kD * kHID, 0, ffn_b2 + (long long)l * kD,
         q_f, t1_f, nullptr, 0, (int)QTOK, kD, kHID, 0, 1);
    ln((int)QTOK, t1_f, ffn_g + (long long)l * kD, ffn_bb + (long long)l * kD, q_f, q_bf, 0);
  }

  // ---- heads ----
  bf16* qn_bf = qw_bf;  // reuse
  ln((int)QTOK, q_f, outln_g, outln_b, nullptr, qn_bf, 0);
  gemm(qn_bf, 0, wb_c1, 0, cls_b1, nullptr, nullptr, h1_bf, 0, (int)QTOK, kD, kD, 1, 1);
  gemm(h1_bf, 0, wb_c2, 0, cls_b2, nullptr, labels_pad, nullptr, 0, (int)QTOK, kNCLS, kD, 0, 1);
  gemm(qn_bf, 0, wb_s1, 0, score_b1, nullptr, nullptr, h1_bf, 0, (int)QTOK, kD, kD, 1, 1);
  gemm(h1_bf, 0, wb_s2, 0, score_b2, nullptr, scores_pad, nullptr, 0, (int)QTOK, 1, kD, 0, 1);
  gemm(qn_bf, (long long)kNQP * kD, mfe_bf, (long long)kN * kD, nullptr, nullptr,
       masks_pad, nullptr, (long long)kNQP * kN, kNQP, kN, kD, 0, kB);

  const long long TOTAL = (long long)kB * kNQ * kNCLS + (long long)kB * kNQ + (long long)kB * kNQ * kN;
  pack_k<<<dim3((unsigned)((TOTAL + 255) / 256)), dim3(256), 0, stream>>>(
      labels_pad, scores_pad, masks_pad, (float*)d_out);
}
